// Attention_74706661146631
// MI455X (gfx1250) — matrix operand order repaired and checked by reading, not compiled, not run on hardware
//
#include <hip/hip_runtime.h>
#include <hip/hip_bf16.h>

typedef _Float16 v16h __attribute__((ext_vector_type(16)));
typedef _Float16 v8h  __attribute__((ext_vector_type(8)));
typedef float    v8f  __attribute__((ext_vector_type(8)));
typedef int      v4i  __attribute__((ext_vector_type(4)));

#define D_MODEL 1024
#define HEADS   16
#define DK      64
#define BATCH   4
#define SEQ     2048
#define MTOT    (BATCH*SEQ)   // 8192

#ifndef __has_builtin
#define __has_builtin(x) 0
#endif
#if __has_builtin(__builtin_amdgcn_global_load_async_to_lds_b128)
#define HAVE_ASYNC_LDS 1
#else
#define HAVE_ASYNC_LDS 0
#endif

#if HAVE_ASYNC_LDS
// 16B global -> LDS async copy (ASYNCcnt-tracked, per CDNA5 ISA).
typedef __attribute__((address_space(1))) v4i* gas_v4i;
typedef __attribute__((address_space(3))) v4i* las_v4i;
__device__ inline void async_copy_b128(const _Float16* g, _Float16* l) {
    __builtin_amdgcn_global_load_async_to_lds_b128((gas_v4i)g, (las_v4i)l, 0, 0);
}
#define ASYNC_WAIT() asm volatile("s_wait_asynccnt 0x0" ::: "memory")
#else
#define ASYNC_WAIT()
#endif

// ---------------- WMMA fragment helpers (wave32, 16x16x32 f16) ----------------
__device__ inline v16h load_a_frag(const _Float16* __restrict__ base, int ld, int koff, int lane) {
    int m = lane & 15, h = (lane >> 4) & 1;
    const _Float16* r = base + (size_t)m * ld + koff;
    return *(const v16h*)(r + h * 16);
}

__device__ inline v16h load_b_frag(const _Float16* __restrict__ bt, int ld, int koff, int lane) {
    int n = lane & 15, h = (lane >> 4) & 1;
    return *(const v16h*)(bt + (size_t)n * ld + koff + h * 16);
}

// ---------------- pack kernels ----------------
__global__ void cvt_f16(const float* __restrict__ in, _Float16* __restrict__ out, int n) {
    int i = blockIdx.x * blockDim.x + threadIdx.x;
    int stride = gridDim.x * blockDim.x;
    for (; i < n; i += stride) out[i] = (_Float16)in[i];
}

__global__ void transpose_cvt(const float* __restrict__ W, _Float16* __restrict__ Wt) {
    int i = blockIdx.x * blockDim.x + threadIdx.x;
    int k = i >> 10, n = i & 1023;
    Wt[((size_t)n << 10) + k] = (_Float16)W[i];
}

// ---------------- GEMM: out = A(f16)[M,1024] @ B + bias, B given as Bt[1024,1024] ----------
// Block tile 128x128, 8 waves, double-buffered Bt tile in LDS, async staging.
// MODE 0: scatter f16 into [B,H,T,DK]   MODE 1: f32 row-major [M,N]
template <int MODE>
__global__ __launch_bounds__(256) void gemm_f16(
    const _Float16* __restrict__ A, const _Float16* __restrict__ Bt,
    const float* __restrict__ bias, _Float16* __restrict__ out16,
    float* __restrict__ outf)
{
    constexpr int K = D_MODEL, N = D_MODEL;
    __shared__ _Float16 bts[2][128 * 32];   // 2 x [128 n][32 k] = 16 KB
    const int tid = threadIdx.x;
    const int wave = tid >> 5, lane = tid & 31;
    const int lanelo = lane & 15, half = (lane >> 4) & 1;
    const int m0 = blockIdx.y * 128;
    const int n0 = blockIdx.x * 128;

    // staging: 128*32 halfs / 256 threads = 16 halfs (32B) per thread
    const int srow = tid >> 1, scol = (tid & 1) * 16;
    const _Float16* gsrc = &Bt[(size_t)(n0 + srow) * K + scol];
    const int loff = srow * 32 + scol;
    const _Float16* arow = A + (size_t)(m0 + wave * 16) * K;

    v8f acc[8] = {};

    auto stage = [&](int buf, int k0) {
#if HAVE_ASYNC_LDS
        async_copy_b128(gsrc + k0,     &bts[buf][loff]);
        async_copy_b128(gsrc + k0 + 8, &bts[buf][loff + 8]);
#else
        *(v16h*)&bts[buf][loff] = *(const v16h*)(gsrc + k0);
#endif
    };
    auto compute = [&](int buf, int k0) {
        v16h a = load_a_frag(arow, K, k0, lane);
#pragma unroll
        for (int nt = 0; nt < 8; ++nt) {
            v16h b = load_b_frag(&bts[buf][nt * 16 * 32], 32, 0, lane);
            acc[nt] = __builtin_amdgcn_wmma_f32_16x16x32_f16(
                false, a, false, b, (short)0, acc[nt], false, false);
        }
    };

    // prologue: stage tile 0
    stage(0, 0);
    ASYNC_WAIT();
    __syncthreads();

    int cur = 0;
    for (int k0 = 0; k0 < K - 32; k0 += 32) {   // stage next unconditionally, no branches
        stage(cur ^ 1, k0 + 32);
        compute(cur, k0);
        ASYNC_WAIT();
        __syncthreads();
        cur ^= 1;
    }
    compute(cur, K - 32);                       // peeled last tile

    // epilogue: C layout -> row = half*8+v, col = nt*16+lanelo
#pragma unroll
    for (int nt = 0; nt < 8; ++nt) {
        int c = n0 + nt * 16 + lanelo;
        float bi = bias[c];
#pragma unroll
        for (int v = 0; v < 8; ++v) {
            int r = m0 + wave * 16 + half * 8 + v;
            float val = acc[nt][v] + bi;
            if (MODE == 0) {
                int b_ = r >> 11, t = r & (SEQ - 1);
                int h_ = c >> 6,  d = c & (DK - 1);
                out16[(((size_t)b_ * HEADS + h_) * SEQ + t) * DK + d] = (_Float16)val;
            } else {
                outf[(size_t)r * N + c] = val;
            }
        }
    }
}

// ---------------- flash attention: one (b,h, 128 q-rows) per workgroup ----------------
__global__ __launch_bounds__(256) void attn_kernel(
    const _Float16* __restrict__ q16, const _Float16* __restrict__ k16,
    const _Float16* __restrict__ v16, _Float16* __restrict__ o16 /*[B,T,C] f16*/)
{
    extern __shared__ _Float16 smem[];
    _Float16* Ks  = smem;               // [128 key][64 d]  (= Bt for Q@K^T)
    _Float16* Vts = smem + 128 * 64;    // [64 d][128 key]  (= Bt for P@V)
    _Float16* Ps  = smem + 2 * 128 * 64;// per-wave [16][128] A-scratch
    const int tid = threadIdx.x;
    const int wave = tid >> 5, lane = tid & 31;
    const int lanelo = lane & 15, half = (lane >> 4) & 1;
    const int bh = blockIdx.y;          // b*16 + h
    const int q0 = blockIdx.x * 128;
    const size_t headbase = (size_t)bh * SEQ * DK;
    _Float16* Pw = Ps + wave * 16 * 128;

    // Q fragments, 1/sqrt(dk)=0.125 folded in
    v16h aq[2];
#pragma unroll
    for (int s = 0; s < 2; ++s) {
        aq[s] = load_a_frag(q16 + headbase + (size_t)(q0 + wave * 16) * DK, DK, s * 32, lane);
#pragma unroll
        for (int i = 0; i < 16; ++i) aq[s][i] = aq[s][i] * (_Float16)0.125f;
    }

    float mstat[8], lstat[8];
    v8f o[4] = {};
    v8f vzero = {};
#pragma unroll
    for (int v = 0; v < 8; ++v) { mstat[v] = -1e30f; lstat[v] = 0.f; }

    for (int kb = 0; kb < SEQ; kb += 128) {
        {   // stage K row-major; stage V transposed [d][key]
            int rr = tid >> 1, c0 = (tid & 1) * 32;
            const _Float16* kg = k16 + headbase + (size_t)(kb + rr) * DK + c0;
            *(v16h*)&Ks[rr * 64 + c0]      = *(const v16h*)kg;
            *(v16h*)&Ks[rr * 64 + c0 + 16] = *(const v16h*)(kg + 16);
            const _Float16* vg = v16 + headbase + (size_t)(kb + rr) * DK + c0;
            v16h t0 = *(const v16h*)vg;
            v16h t1 = *(const v16h*)(vg + 16);
#pragma unroll
            for (int i = 0; i < 16; ++i) {
                Vts[(c0 + i) * 128 + rr]      = t0[i];
                Vts[(c0 + 16 + i) * 128 + rr] = t1[i];
            }
            if (kb + 128 < SEQ) {       // prefetch next K/V block (global_prefetch_b8)
                __builtin_prefetch(kg + 128 * DK, 0, 0);
                __builtin_prefetch(vg + 128 * DK, 0, 0);
            }
        }
        __syncthreads();

        // S = (Q*scale) @ K^T : 8 n-tiles of 16 keys, K-dim 64 = 2 wmma steps
        v8f sacc[8];
#pragma unroll
        for (int nt = 0; nt < 8; ++nt) {
            sacc[nt] = vzero;
#pragma unroll
            for (int s = 0; s < 2; ++s) {
                v16h b = load_b_frag(Ks + nt * 16 * 64, 64, s * 32, lane);
                sacc[nt] = __builtin_amdgcn_wmma_f32_16x16x32_f16(
                    false, aq[s], false, b, (short)0, sacc[nt], false, false);
            }
        }
        // online softmax; row r = half*8+v lives in one 16-lane half -> width-16 shuffles
        float newm[8], fac[8];
#pragma unroll
        for (int v = 0; v < 8; ++v) {
            float rm = sacc[0][v];
#pragma unroll
            for (int nt = 1; nt < 8; ++nt) rm = fmaxf(rm, sacc[nt][v]);
#pragma unroll
            for (int off = 1; off < 16; off <<= 1) rm = fmaxf(rm, __shfl_xor(rm, off, 16));
            newm[v] = fmaxf(mstat[v], rm);
            fac[v]  = __expf(mstat[v] - newm[v]);
        }
#pragma unroll
        for (int v = 0; v < 8; ++v) {
            float rs = 0.f;
#pragma unroll
            for (int nt = 0; nt < 8; ++nt) {
                float p = __expf(sacc[nt][v] - newm[v]);
                rs += p;
                Pw[(half * 8 + v) * 128 + nt * 16 + lanelo] = (_Float16)p;  // C-layout -> A-scratch
            }
#pragma unroll
            for (int off = 1; off < 16; off <<= 1) rs += __shfl_xor(rs, off, 16);
            lstat[v] = lstat[v] * fac[v] + rs;
            mstat[v] = newm[v];
        }
#pragma unroll
        for (int nt = 0; nt < 4; ++nt)
#pragma unroll
            for (int v = 0; v < 8; ++v) o[nt][v] *= fac[v];

        // O += P @ V : K-dim 128 = 4 wmma steps, 4 d-tiles (same-wave LDS is in-order)
#pragma unroll
        for (int s = 0; s < 4; ++s) {
            v16h ap = load_a_frag(Pw, 128, s * 32, lane);
#pragma unroll
            for (int nt = 0; nt < 4; ++nt) {
                v16h bv = load_b_frag(Vts + nt * 16 * 128, 128, s * 32, lane);
                o[nt] = __builtin_amdgcn_wmma_f32_16x16x32_f16(
                    false, ap, false, bv, (short)0, o[nt], false, false);
            }
        }
        __syncthreads();
    }
    // normalize, scatter to [B,T,C] f16 for the Wo GEMM
    const int b_ = bh >> 4, h_ = bh & 15;
#pragma unroll
    for (int v = 0; v < 8; ++v) {
        float inv = 1.f / lstat[v];
        int tok = q0 + wave * 16 + half * 8 + v;
        size_t rowbase = ((size_t)b_ * SEQ + tok) * D_MODEL + h_ * DK;
#pragma unroll
        for (int nt = 0; nt < 4; ++nt)
            o16[rowbase + nt * 16 + lanelo] = (_Float16)(o[nt][v] * inv);
    }
}

// ---------------- launch ----------------
extern "C" void kernel_launch(void* const* d_in, const int* in_sizes, int n_in,
                              void* d_out, int out_size, void* d_ws, size_t ws_size,
                              hipStream_t stream) {
    const float* x  = (const float*)d_in[0];
    const float* Wq = (const float*)d_in[1];
    const float* bq = (const float*)d_in[2];
    const float* Wk = (const float*)d_in[3];
    const float* bk = (const float*)d_in[4];
    const float* Wv = (const float*)d_in[5];
    const float* bv = (const float*)d_in[6];
    const float* Wo = (const float*)d_in[7];
    const float* bo = (const float*)d_in[8];
    float* out = (float*)d_out;

    char* ws = (char*)d_ws;
    const size_t MB = 1u << 20;
    _Float16* x16 = (_Float16*)(ws);             // 16 MB
    _Float16* Wqt = (_Float16*)(ws + 16 * MB);   //  2 MB each
    _Float16* Wkt = (_Float16*)(ws + 18 * MB);
    _Float16* Wvt = (_Float16*)(ws + 20 * MB);
    _Float16* Wot = (_Float16*)(ws + 22 * MB);
    _Float16* q16 = (_Float16*)(ws + 24 * MB);   // 16 MB each [B,H,T,DK]
    _Float16* k16 = (_Float16*)(ws + 40 * MB);
    _Float16* v16 = (_Float16*)(ws + 56 * MB);
    _Float16* a16 = (_Float16*)(ws + 72 * MB);   // 16 MB [B,T,C]   (total 88 MB)

    cvt_f16<<<2048, 256, 0, stream>>>(x, x16, MTOT * D_MODEL);
    transpose_cvt<<<4096, 256, 0, stream>>>(Wq, Wqt);
    transpose_cvt<<<4096, 256, 0, stream>>>(Wk, Wkt);
    transpose_cvt<<<4096, 256, 0, stream>>>(Wv, Wvt);
    transpose_cvt<<<4096, 256, 0, stream>>>(Wo, Wot);

    dim3 ggrid(D_MODEL / 128, MTOT / 128);
    gemm_f16<0><<<ggrid, 256, 0, stream>>>(x16, Wqt, bq, q16, nullptr);
    gemm_f16<0><<<ggrid, 256, 0, stream>>>(x16, Wkt, bk, k16, nullptr);
    gemm_f16<0><<<ggrid, 256, 0, stream>>>(x16, Wvt, bv, v16, nullptr);

    attn_kernel<<<dim3(SEQ / 128, BATCH * HEADS), 256, 64 * 1024, stream>>>(q16, k16, v16, a16);

    gemm_f16<1><<<ggrid, 256, 0, stream>>>(a16, Wot, bo, nullptr, out);
}
